// LSTMDecoder_14920716387071
// MI455X (gfx1250) — matrix operand order repaired and checked by reading, not compiled, not run on hardware
//
#include <hip/hip_runtime.h>
#include <hip/hip_bf16.h>

typedef __attribute__((ext_vector_type(16))) __bf16 v16bf;
typedef __attribute__((ext_vector_type(8)))  __bf16 v8bf;
typedef __attribute__((ext_vector_type(8)))  float  v8f;

#define B_  64
#define T_  256
#define S_  256
#define D_  1024

__device__ __forceinline__ float sigf(float x) { return 1.f / (1.f + __expf(-x)); }

// ---------------------------------------------------------------------------
// Pack weights W[N,K] (optionally two concatenated along K, zero padded) into
// WMMA B-fragment tiles: Bp[kt][nt][lane][e], where for lane L, element e:
//   n = nt*16 + (L&15),  k = kt*32 + 16*(e>=8) + 8*(L>=16) + (e&7)
// so a wave's fragment load is one contiguous 32B read per lane.
// ---------------------------------------------------------------------------
__global__ void pack_b(const float* __restrict__ W0, int K0,
                       const float* __restrict__ W1, int K1,
                       __bf16* __restrict__ Bp, int N, int Ktiles) {
    int Ntiles = N >> 4;
    long total = (long)Ktiles * Ntiles * 512;
    long idx = (long)blockIdx.x * blockDim.x + threadIdx.x;
    if (idx >= total) return;
    int e    = (int)(idx & 15);
    int lane = (int)((idx >> 4) & 31);
    long tile = idx >> 9;
    int nt = (int)(tile % Ntiles);
    int kt = (int)(tile / Ntiles);
    int n = nt * 16 + (lane & 15);
    int k = kt * 32 + ((e >> 3) << 4) + ((lane >> 4) << 3) + (e & 7);
    float v = 0.f;
    if (k < K0)            v = W0[(long)n * K0 + k];
    else if (k < K0 + K1)  v = W1[(long)n * K1 + (k - K0)];
    Bp[idx] = (__bf16)v;
}

__global__ void f32_to_bf16_kernel(const float* __restrict__ in,
                                   __bf16* __restrict__ out, long n) {
    long i = (long)blockIdx.x * blockDim.x + threadIdx.x;
    if (i < n) out[i] = (__bf16)in[i];
}

union ABfrag { v16bf v; v8bf h[2]; };

// ---------------------------------------------------------------------------
// Generic bf16 WMMA GEMM: C = A[M,Kpad] * B(packed), fp32 accumulate.
// Each wave computes a 32x64 output tile: 2 A fragments x 4 B fragments,
// 8 WMMAs per K step. Explicit double buffering with pure pointer-increment
// addressing: next-iteration fragments are always prefetched (the final,
// unused prefetch over-reads <=64B past A and Ntiles*1KB past B into the next 256B-aligned ws buffer,
// which is safe), so the K loop has no per-iteration address multiplies and
// the waits in front of the WMMAs stay partial.
// A fragment (16-bit A 16x32 layout): lane<16 row gets K{0..7,16..23};
// lane>=16 gets K{8..15,24..31}.
// PROJT epilogue stores transposed bf16 into projT[b][d][s] (row = b*256+s).
// ---------------------------------------------------------------------------
template <bool PROJT>
__global__ void wmma_gemm(const __bf16* __restrict__ A, int Kpad,
                          const __bf16* __restrict__ Bp,
                          float* __restrict__ C, __bf16* __restrict__ Ct,
                          int Mgroups, int N, int Ktiles) {
    int Ntiles = N >> 4;
    int nsuper = N >> 6;
    int lane = threadIdx.x & 31;
    int w = (blockIdx.x * blockDim.x + threadIdx.x) >> 5;
    if (w >= Mgroups * nsuper) return;
    int mg = w % Mgroups;
    int ng = w / Mgroups;

    int r0 = mg * 32 + (lane & 15);
    int koff = (lane >> 4) << 3;  // 0 or 8
    const __bf16* ap0 = A + (long)r0 * Kpad + koff;
    const __bf16* ap1 = ap0 + (long)16 * Kpad;
    const __bf16* bp  = Bp + ((long)(ng << 2) << 9) + (long)lane * 16;
    const long bstride = (long)Ntiles << 9;

    v8f acc00 = {}, acc01 = {}, acc02 = {}, acc03 = {};
    v8f acc10 = {}, acc11 = {}, acc12 = {}, acc13 = {};

    // prologue: fragments for kt = 0
    ABfrag a0c, a1c;
    a0c.h[0] = *(const v8bf*)(ap0);  a0c.h[1] = *(const v8bf*)(ap0 + 16);
    a1c.h[0] = *(const v8bf*)(ap1);  a1c.h[1] = *(const v8bf*)(ap1 + 16);
    v16bf b0c = *(const v16bf*)(bp);
    v16bf b1c = *(const v16bf*)(bp + 512);
    v16bf b2c = *(const v16bf*)(bp + 1024);
    v16bf b3c = *(const v16bf*)(bp + 1536);
    ap0 += 32; ap1 += 32; bp += bstride;

    for (int kt = 0; kt < Ktiles; ++kt) {
        // issue next-iteration loads first (double buffer, pointer increments)
        ABfrag a0n, a1n;
        a0n.h[0] = *(const v8bf*)(ap0);  a0n.h[1] = *(const v8bf*)(ap0 + 16);
        a1n.h[0] = *(const v8bf*)(ap1);  a1n.h[1] = *(const v8bf*)(ap1 + 16);
        v16bf b0n = *(const v16bf*)(bp);
        v16bf b1n = *(const v16bf*)(bp + 512);
        v16bf b2n = *(const v16bf*)(bp + 1024);
        v16bf b3n = *(const v16bf*)(bp + 1536);
        ap0 += 32; ap1 += 32; bp += bstride;

        // compute on current fragments while next loads are in flight
        acc00 = __builtin_amdgcn_wmma_f32_16x16x32_bf16(false, a0c.v, false, b0c, (short)0, acc00, false, false);
        acc10 = __builtin_amdgcn_wmma_f32_16x16x32_bf16(false, a1c.v, false, b0c, (short)0, acc10, false, false);
        acc01 = __builtin_amdgcn_wmma_f32_16x16x32_bf16(false, a0c.v, false, b1c, (short)0, acc01, false, false);
        acc11 = __builtin_amdgcn_wmma_f32_16x16x32_bf16(false, a1c.v, false, b1c, (short)0, acc11, false, false);
        acc02 = __builtin_amdgcn_wmma_f32_16x16x32_bf16(false, a0c.v, false, b2c, (short)0, acc02, false, false);
        acc12 = __builtin_amdgcn_wmma_f32_16x16x32_bf16(false, a1c.v, false, b2c, (short)0, acc12, false, false);
        acc03 = __builtin_amdgcn_wmma_f32_16x16x32_bf16(false, a0c.v, false, b3c, (short)0, acc03, false, false);
        acc13 = __builtin_amdgcn_wmma_f32_16x16x32_bf16(false, a1c.v, false, b3c, (short)0, acc13, false, false);

        a0c = a0n; a1c = a1n;
        b0c = b0n; b1c = b1n; b2c = b2n; b3c = b3n;
    }

    // C layout: VGPR e -> row = tile_row + e + 8*(lane>=16), col = nt*16 + (lane&15)
    int rb = mg * 32 + ((lane >> 4) << 3);
    int cb = (ng << 6) + (lane & 15);
#pragma unroll
    for (int m = 0; m < 2; ++m) {
#pragma unroll
        for (int j = 0; j < 4; ++j) {
            v8f acc;
            if (m == 0) acc = (j == 0) ? acc00 : (j == 1) ? acc01 : (j == 2) ? acc02 : acc03;
            else        acc = (j == 0) ? acc10 : (j == 1) ? acc11 : (j == 2) ? acc12 : acc13;
            int col = cb + (j << 4);
            int rbase = rb + (m << 4);
#pragma unroll
            for (int e = 0; e < 8; ++e) {
                int r = rbase + e;
                if (!PROJT) {
                    C[(long)r * N + col] = acc[e];
                } else {
                    int b = r >> 8, s = r & 255;
                    Ct[(((long)b * D_ + col) << 8) + s] = (__bf16)acc[e];
                }
            }
        }
    }
}

// ---------------------------------------------------------------------------
// LSTM pointwise: gates(+biases) -> c,h ; stage h as bf16 for next-step gates
// GEMM (cols 1040..2063) and this step's feed GEMM (cols 0..1023).
// ---------------------------------------------------------------------------
__global__ void lstm_pointwise(const float* __restrict__ gates,
                               const float* __restrict__ b_ih, const float* __restrict__ b_hh,
                               float* __restrict__ c, float* __restrict__ h,
                               __bf16* __restrict__ gatesA, __bf16* __restrict__ feedA) {
    int idx = blockIdx.x * blockDim.x + threadIdx.x;  // 65536
    int b = idx >> 10, d = idx & 1023;
    const float* g = gates + ((long)b << 12);
    float gi = g[d]          + b_ih[d]          + b_hh[d];
    float gf = g[D_ + d]     + b_ih[D_ + d]     + b_hh[D_ + d];
    float gg = g[2 * D_ + d] + b_ih[2 * D_ + d] + b_hh[2 * D_ + d];
    float go = g[3 * D_ + d] + b_ih[3 * D_ + d] + b_hh[3 * D_ + d];
    float cn = sigf(gf) * c[idx] + sigf(gi) * tanhf(gg);
    float hn = sigf(go) * tanhf(cn);
    c[idx] = cn;
    h[idx] = hn;
    gatesA[(long)b * 2080 + 1040 + d] = (__bf16)hn;
    feedA[((long)b << 11) + d] = (__bf16)hn;
}

// ---------------------------------------------------------------------------
// Attention: scores = h . projT (coalesced in s), masked softmax, context,
// write attn weights to output, stage ctx bf16 into feed activations.
// ---------------------------------------------------------------------------
__global__ void attn_kernel(const float* __restrict__ h, const __bf16* __restrict__ projT,
                            const __bf16* __restrict__ enc_bf,
                            const unsigned char* __restrict__ mask,
                            float* __restrict__ attn_out,
                            __bf16* __restrict__ feedA, int t) {
    __shared__ float sh[D_];
    __shared__ float sw[S_];
    __shared__ float red[256];
    int b = blockIdx.x, tid = threadIdx.x;
    for (int i = tid; i < D_; i += 256) sh[i] = h[(b << 10) + i];
    __syncthreads();

    const __bf16* pT = projT + (((long)b * D_) << 8) + tid;  // [d][s], s = tid
    float acc = 0.f;
#pragma unroll 4
    for (int d = 0; d < D_; ++d) acc = fmaf(sh[d], (float)pT[(long)d << 8], acc);

    bool pad = mask[(b << 8) + tid] != 0;
    float score = pad ? -3.402823466e38f : acc;
    red[tid] = score; __syncthreads();
    for (int off = 128; off > 0; off >>= 1) {
        if (tid < off) red[tid] = fmaxf(red[tid], red[tid + off]);
        __syncthreads();
    }
    float m = red[0]; __syncthreads();
    float ex = pad ? 0.f : __expf(score - m);
    red[tid] = ex; __syncthreads();
    for (int off = 128; off > 0; off >>= 1) {
        if (tid < off) red[tid] += red[tid + off];
        __syncthreads();
    }
    float wgt = ex / red[0];
    sw[tid] = wgt;
    attn_out[(((long)b * T_ + t) << 8) + tid] = wgt;
    __syncthreads();

    float c0 = 0, c1 = 0, c2 = 0, c3 = 0;
    const __bf16* eb = enc_bf + ((long)b << 18);
    for (int s = 0; s < S_; ++s) {
        float wv = sw[s];
        const __bf16* row = eb + (s << 10);
        c0 = fmaf(wv, (float)row[tid], c0);
        c1 = fmaf(wv, (float)row[tid + 256], c1);
        c2 = fmaf(wv, (float)row[tid + 512], c2);
        c3 = fmaf(wv, (float)row[tid + 768], c3);
    }
    __bf16* fA = feedA + ((long)b << 11) + D_;
    fA[tid] = (__bf16)c0;
    fA[tid + 256] = (__bf16)c1;
    fA[tid + 512] = (__bf16)c2;
    fA[tid + 768] = (__bf16)c3;
}

// ---------------------------------------------------------------------------
// feed = tanh(feedpre); stage feed bf16 for next gates GEMM; tiny out proj
// (4 dots of 1024 via block reduce) with tanh/relu; stage x_{t+1}.
// ---------------------------------------------------------------------------
__global__ void feed_out_kernel(const float* __restrict__ feedpre,
                                const float* __restrict__ W_fin, const float* __restrict__ b_fin,
                                const float* __restrict__ tgt,
                                __bf16* __restrict__ gatesA,
                                float* __restrict__ out, int t) {
    __shared__ float r0[256], r1[256], r2[256], r3[256];
    int b = blockIdx.x, tid = threadIdx.x;
    float a0 = 0, a1 = 0, a2 = 0, a3 = 0;
#pragma unroll
    for (int i = 0; i < 4; ++i) {
        int d = tid + (i << 8);
        float f = tanhf(feedpre[(b << 10) + d]);
        gatesA[(long)b * 2080 + 16 + d] = (__bf16)f;
        a0 = fmaf(f, W_fin[d], a0);
        a1 = fmaf(f, W_fin[D_ + d], a1);
        a2 = fmaf(f, W_fin[2 * D_ + d], a2);
        a3 = fmaf(f, W_fin[3 * D_ + d], a3);
    }
    r0[tid] = a0; r1[tid] = a1; r2[tid] = a2; r3[tid] = a3;
    __syncthreads();
    for (int off = 128; off > 0; off >>= 1) {
        if (tid < off) {
            r0[tid] += r0[tid + off]; r1[tid] += r1[tid + off];
            r2[tid] += r2[tid + off]; r3[tid] += r3[tid + off];
        }
        __syncthreads();
    }
    if (tid < 16 && (t + 1) < T_)
        gatesA[(long)b * 2080 + tid] = (__bf16)tgt[((long)b * T_ + (t + 1)) * 16 + tid];
    if (tid == 0) {
        long o = ((long)b * T_ + t) << 2;
        out[o + 0] = tanhf(r0[0] + b_fin[0]);
        out[o + 1] = tanhf(r1[0] + b_fin[1]);
        out[o + 2] = tanhf(r2[0] + b_fin[2]);
        out[o + 3] = fmaxf(0.f, r3[0] + b_fin[3]);
    }
}

// Initial staging: gatesA = bf16 concat(x_0, feed0, h0, pad); c = c0.
__global__ void init_kernel(const float* __restrict__ tgt, const float* __restrict__ feed0,
                            const float* __restrict__ h0, const float* __restrict__ c0,
                            __bf16* __restrict__ gatesA, float* __restrict__ c) {
    long idx = (long)blockIdx.x * blockDim.x + threadIdx.x;
    long nA = (long)B_ * 2080;
    if (idx < nA) {
        int b = (int)(idx / 2080), kcol = (int)(idx % 2080);
        float v = 0.f;
        if (kcol < 16)        v = tgt[((long)b * T_) * 16 + kcol];
        else if (kcol < 1040) v = feed0[(b << 10) + kcol - 16];
        else if (kcol < 2064) v = h0[(b << 10) + kcol - 1040];
        gatesA[idx] = (__bf16)v;
    } else if (idx < nA + (long)B_ * D_) {
        long i = idx - nA;
        c[i] = c0[i];
    }
}

extern "C" void kernel_launch(void* const* d_in, const int* in_sizes, int n_in,
                              void* d_out, int out_size, void* d_ws, size_t ws_size,
                              hipStream_t stream) {
    const float* tgt   = (const float*)d_in[0];
    const float* enc   = (const float*)d_in[1];
    const unsigned char* mask = (const unsigned char*)d_in[2];
    const float* W_ih  = (const float*)d_in[3];
    const float* W_hh  = (const float*)d_in[4];
    const float* b_ih  = (const float*)d_in[5];
    const float* b_hh  = (const float*)d_in[6];
    const float* W_src = (const float*)d_in[7];
    const float* W_cph = (const float*)d_in[8];
    const float* W_fin = (const float*)d_in[9];
    const float* b_fin = (const float*)d_in[10];
    const float* h0    = (const float*)d_in[11];
    const float* c0    = (const float*)d_in[12];
    const float* feed0 = (const float*)d_in[13];

    float* out      = (float*)d_out;
    float* attn_out = out + (long)B_ * T_ * 4;

    char* ws = (char*)d_ws;
    size_t off = 0;
    auto alloc = [&](size_t bytes) -> char* {
        char* p = ws + off;
        off = (off + bytes + 255) & ~(size_t)255;
        return p;
    };
    __bf16* Bp_g    = (__bf16*)alloc(65UL * 256 * 512 * 2);   // gates weights (W_ih|W_hh, K padded to 2080)
    __bf16* Bp_cph  = (__bf16*)alloc(64UL * 64 * 512 * 2);    // W_cph
    __bf16* Bp_src  = (__bf16*)alloc(32UL * 64 * 512 * 2);    // W_src
    __bf16* enc_bf  = (__bf16*)alloc(16777216UL * 2);         // enc bf16 [B,S,E]
    __bf16* projT   = (__bf16*)alloc(16777216UL * 2);         // projT bf16 [B,D,S]
    __bf16* gatesA  = (__bf16*)alloc(64UL * 2080 * 2);        // [x|feed|h|pad]
    __bf16* feedA   = (__bf16*)alloc(64UL * 2048 * 2);        // [h|ctx]
    float*  gates   = (float*)alloc(64UL * 4096 * 4);
    float*  feedpre = (float*)alloc(64UL * 1024 * 4);
    float*  hbuf    = (float*)alloc(64UL * 1024 * 4);
    float*  cbuf    = (float*)alloc(64UL * 1024 * 4);
    (void)alloc(4096);  // tail pad: final double-buffer over-prefetch stays in ws

    // ---- one-time prep (re-done every call for determinism) ----
    { long tot = 65L * 256 * 512;
      pack_b<<<(tot + 255) / 256, 256, 0, stream>>>(W_ih, 1040, W_hh, 1024, Bp_g, 4096, 65); }
    { long tot = 64L * 64 * 512;
      pack_b<<<(tot + 255) / 256, 256, 0, stream>>>(W_cph, 2048, nullptr, 0, Bp_cph, 1024, 64); }
    { long tot = 32L * 64 * 512;
      pack_b<<<(tot + 255) / 256, 256, 0, stream>>>(W_src, 1024, nullptr, 0, Bp_src, 1024, 32); }
    f32_to_bf16_kernel<<<65536, 256, 0, stream>>>(enc, enc_bf, 16777216L);
    // proj_enc GEMM: M=16384 (Mgroups=512), N=1024, K=1024; transposing bf16 epilogue
    wmma_gemm<true><<<2048, 128, 0, stream>>>(enc_bf, 1024, Bp_src, nullptr, projT, 512, 1024, 32);
    { long tot = 64L * 2080 + 65536;
      init_kernel<<<(tot + 255) / 256, 256, 0, stream>>>(tgt, feed0, h0, c0, gatesA, cbuf); }

    // ---- sequential decode ----
    for (int t = 0; t < T_; ++t) {
        // gates = [x_t|feed|h] @ [W_ih|W_hh]^T : M=64 (Mgroups=2), N=4096, K=2080
        // 128 waves in 128-thread blocks -> spread over 32 WGPs
        wmma_gemm<false><<<32, 128, 0, stream>>>(gatesA, 2080, Bp_g, gates, nullptr, 2, 4096, 65);
        lstm_pointwise<<<256, 256, 0, stream>>>(gates, b_ih, b_hh, cbuf, hbuf, gatesA, feedA);
        attn_kernel<<<64, 256, 0, stream>>>(hbuf, projT, enc_bf, mask, attn_out, feedA, t);
        // feed_pre = [h|ctx] @ W_cph^T : M=64 (Mgroups=2), N=1024, K=2048
        wmma_gemm<false><<<8, 128, 0, stream>>>(feedA, 2048, Bp_cph, feedpre, nullptr, 2, 1024, 64);
        feed_out_kernel<<<64, 256, 0, stream>>>(feedpre, W_fin, b_fin, tgt, gatesA, out, t);
    }
}
